// AttentionTSSA_23493471109512
// MI455X (gfx1250) — matrix operand order repaired and checked by reading, not compiled, not run on hardware
//
#include <hip/hip_runtime.h>
#include <hip/hip_bf16.h>
#include <stdint.h>

// Problem constants (match reference)
#define B_  8
#define N_  8192
#define C_  512
#define H_  8
#define D_  64

typedef __attribute__((ext_vector_type(16))) __bf16 v16bf;
typedef __attribute__((ext_vector_type(8)))  float  v8f;
typedef __attribute__((ext_vector_type(4)))  int    v4i;

#define AS1 __attribute__((address_space(1)))
#define AS3 __attribute__((address_space(3)))

// ---- feature detection (compile-safe fallbacks) ---------------------------
#if defined(__has_builtin)
#  if __has_builtin(__builtin_amdgcn_global_load_async_to_lds_b128)
#    define HAVE_ASYNC 1
#  endif
#endif
#ifndef HAVE_ASYNC
#  define HAVE_ASYNC 0
#endif

#if HAVE_ASYNC
#  if __has_builtin(__builtin_amdgcn_s_wait_asynccnt)
#    define ASYNC_WAIT() __builtin_amdgcn_s_wait_asynccnt(0)
#  else
#    define ASYNC_WAIT() asm volatile("s_wait_asynccnt 0x0" ::: "memory")
#  endif
#else
#  define ASYNC_WAIT() ((void)0)
#endif

union FragBF {
    v16bf v;
    unsigned int u[8];
};

__device__ __forceinline__ unsigned short f2bf(float f) {
    unsigned int u = __float_as_uint(f);
    u += 0x7FFFu + ((u >> 16) & 1u);       // round-to-nearest-even
    return (unsigned short)(u >> 16);
}

#if defined(__has_builtin)
#  if __has_builtin(__builtin_amdgcn_cvt_pk_bf16_f32)
#    define HAVE_PKCVT 1
#  endif
#endif

#if defined(HAVE_PKCVT)
typedef __attribute__((ext_vector_type(2))) __bf16 v2bf;
__device__ __forceinline__ unsigned int pk2bf(float a, float b) {
    union { v2bf v; unsigned int u; } cv;
    cv.v = __builtin_amdgcn_cvt_pk_bf16_f32(a, b);   // D[15:0]=bf16(a), D[31:16]=bf16(b)
    return cv.u;
}
#else
// 3 VALU per pair: +0x8000 round bias, then one v_perm_b32 packs both hi-halves.
// dst[15:0] = a[31:16] (bytes 2,3 of src1), dst[31:16] = b[31:16] (bytes 2,3 of src0)
__device__ __forceinline__ unsigned int pk2bf(float a, float b) {
    const unsigned int ua = __float_as_uint(a) + 0x8000u;
    const unsigned int ub = __float_as_uint(b) + 0x8000u;
    return __builtin_amdgcn_perm(ub, ua, 0x07060302u);
}
#endif

// ---------------------------------------------------------------------------
// K0: fp32 -> bf16 weight conversion, and accumulator zeroing
// ---------------------------------------------------------------------------
__global__ __launch_bounds__(256)
void cvt_bf16_kernel(const float* __restrict__ src, unsigned short* __restrict__ dst, int n) {
    int i = blockIdx.x * 256 + threadIdx.x;
    if (i < n) dst[i] = f2bf(src[i]);
}

__global__ __launch_bounds__(256)
void zero_kernel(float* __restrict__ p, int n) {
    int i = blockIdx.x * 256 + threadIdx.x;
    if (i < n) p[i] = 0.0f;
}

// ---------------------------------------------------------------------------
// K1 / K3: WMMA GEMM  Out[m][c] = (scale? coef*A : A)[m][k] * Wrow[c][k] (+bias)
//   Block tile 128(M) x 64(N), K staged 32-wide, double-buffered LDS.
//   8 waves (4 M x 2 N), each wave: 32x32 tile = 4x v_wmma_f32_16x16x32_bf16
//   per K step.  B tile copied global->LDS with GLOBAL_LOAD_ASYNC_TO_LDS_B128
//   (ASYNCcnt); A tile converted fp32->bf16 in registers (v_perm pack).
// ---------------------------------------------------------------------------
template<bool SCALED>
__global__ __launch_bounds__(256)
void wmma_gemm_kernel(const float* __restrict__ A,
                      const unsigned short* __restrict__ Bw,
                      const float* __restrict__ coef,   // (B*H*N) or null
                      const float* __restrict__ bias,   // (C) or null
                      float* __restrict__ Out)
{
    __shared__ __align__(16) unsigned short As[2][128][36];  // 32 bf16 + pad (72B rows)
    __shared__ __align__(16) unsigned short Bs[2][64][40];   // 80B rows (16B aligned chunks)

    const int tid   = threadIdx.x;
    const int lane  = tid & 31;
    const int wave  = tid >> 5;
    const int wm    = (wave & 3) << 5;
    const int wn    = (wave >> 2) << 5;
    const int mBase = blockIdx.x << 7;   // *128
    const int nBase = blockIdx.y << 6;   // *64
    const int half  = lane >> 4;
    const int l16   = lane & 15;

    v8f acc[2][2];
#pragma unroll
    for (int i = 0; i < 2; i++)
#pragma unroll
        for (int j = 0; j < 2; j++)
#pragma unroll
            for (int e = 0; e < 8; e++) acc[i][j][e] = 0.0f;

    float4 av[4];
    float  sc[4];
#if !HAVE_ASYNC
    uint2  bv[2];
#endif

    // --- staging helpers (fully unrolled, uniform control flow) ------------
    auto issueA = [&](int k0) {
#pragma unroll
        for (int it = 0; it < 4; it++) {
            const int c  = tid + (it << 8);
            const int m  = c >> 3;
            const int kq = (c & 7) << 2;
            const size_t row = (size_t)(mBase + m);
            av[it] = *(const float4*)(A + row * C_ + k0 + kq);
            if (SCALED) {
                const int b = (int)(row >> 13);             // N_ = 8192
                const int n = (int)(row & (N_ - 1));
                const int h = (k0 + kq) >> 6;               // same head for all 4
                sc[it] = coef[(((size_t)b * H_ + h) << 13) + n];
            } else {
                sc[it] = 1.0f;
            }
        }
    };
    auto issueB = [&](int k0, int buf) {
#if HAVE_ASYNC
        // 64 rows x 32 bf16 = 4KB tile; 256 lanes x 16B = one async instruction
        const int n = tid >> 2;
        const int c = tid & 3;
        __builtin_amdgcn_global_load_async_to_lds_b128(
            (AS1 v4i*)(void*)(Bw + (size_t)(nBase + n) * C_ + k0 + (c << 3)),
            (AS3 v4i*)(void*)&Bs[buf][n][c << 3], 0, 0);
#else
#pragma unroll
        for (int it = 0; it < 2; it++) {
            const int c  = tid + (it << 8);
            const int n  = c >> 3;
            const int kq = (c & 7) << 2;
            bv[it] = *(const uint2*)(Bw + (size_t)(nBase + n) * C_ + k0 + kq);
        }
        (void)buf;
#endif
    };
    auto storeAB = [&](int buf) {
#pragma unroll
        for (int it = 0; it < 4; it++) {
            const int c  = tid + (it << 8);
            const int m  = c >> 3;
            const int kq = (c & 7) << 2;
            const float s = sc[it];
            uint2 p;
            p.x = pk2bf(av[it].x * s, av[it].y * s);
            p.y = pk2bf(av[it].z * s, av[it].w * s);
            *(uint2*)&As[buf][m][kq] = p;
        }
#if !HAVE_ASYNC
#pragma unroll
        for (int it = 0; it < 2; it++) {
            const int c  = tid + (it << 8);
            const int n  = c >> 3;
            const int kq = (c & 7) << 2;
            *(uint2*)&Bs[buf][n][kq] = bv[it];
        }
#endif
    };

    // --- prologue: fill buffer 0 -------------------------------------------
    issueA(0);
    issueB(0, 0);
    storeAB(0);
    ASYNC_WAIT();
    __syncthreads();

    int buf = 0;
    for (int k0 = 0; k0 < C_; k0 += 32) {
        const bool hasNext = (k0 + 32) < C_;
        if (hasNext) {
            issueA(k0 + 32);          // global loads in flight during compute
            issueB(k0 + 32, buf ^ 1); // async copy straight into next LDS buffer
        }

        // ---- build fragments per ISA 7.12.2 16-bit layouts from current buf
        FragBF af[2], bf[2];
#pragma unroll
        for (int t = 0; t < 2; t++) {
            // A 16x32: lanes 0-15 -> K 0..7/16..23 ; lanes 16-31 -> K 8..15/24..31
            const int m  = wm + (t << 4) + l16;
            const int kb = half << 3;
#pragma unroll
            for (int j = 0; j < 4; j++)
                af[t].u[j] = *(const unsigned int*)&As[buf][m][kb + (j << 1)];
#pragma unroll
            for (int j = 0; j < 4; j++)
                af[t].u[4 + j] = *(const unsigned int*)&As[buf][m][kb + 16 + (j << 1)];
            // B 32x16: lanes 0-15 -> K 0..7/16..23 ; lanes 16-31 -> K 8..15/24..31
            const int n   = wn + (t << 4) + l16;
            const int kbB = half << 3;
#pragma unroll
            for (int j = 0; j < 8; j++)
                bf[t].u[j] = *(const unsigned int*)&Bs[buf][n][kbB + (j << 1) + ((j >> 2) << 3)];
        }

#pragma unroll
        for (int i = 0; i < 2; i++)
#pragma unroll
            for (int j = 0; j < 2; j++)
                acc[i][j] = __builtin_amdgcn_wmma_f32_16x16x32_bf16(
                    false, af[i].v, false, bf[j].v,
                    (short)0, acc[i][j], false, false);

        if (hasNext) {
            storeAB(buf ^ 1);   // safe: buf^1 last read before previous barrier
            ASYNC_WAIT();       // B tile of buf^1 landed in LDS
            __syncthreads();    // waits DScnt, publishes As/Bs[buf^1]
            buf ^= 1;
        }
    }

    // ---- epilogue: C/D layout: lane -> col, VGPR r -> row (+8 for hi lanes)
#pragma unroll
    for (int i = 0; i < 2; i++) {
#pragma unroll
        for (int j = 0; j < 2; j++) {
            const int col  = nBase + wn + (j << 4) + l16;
            const float bv2 = bias ? bias[col] : 0.0f;
            const int mrow = mBase + wm + (i << 4) + (half << 3);
#pragma unroll
            for (int r = 0; r < 8; r++)
                Out[(size_t)(mrow + r) * C_ + col] = acc[i][j][r] + bv2;
        }
    }
}

// ---------------------------------------------------------------------------
// K2a: colsq[b][c] = sum_n w[b,n,c]^2
// ---------------------------------------------------------------------------
__global__ __launch_bounds__(256)
void colsq_kernel(const float* __restrict__ W, float* __restrict__ colsq)
{
    const int blk = blockIdx.x;                 // B * N/128 = 512 blocks
    const int b   = blk / (N_ / 128);
    const int n0  = (blk % (N_ / 128)) * 128;
    const int tid = threadIdx.x;
    const float* base = W + ((size_t)b * N_ + n0) * C_;
    float a0 = 0.0f, a1 = 0.0f;
    for (int i = 0; i < 128; i++) {
        const float x0 = base[(size_t)i * C_ + tid];
        const float x1 = base[(size_t)i * C_ + tid + 256];
        a0 += x0 * x0;
        a1 += x1 * x1;
    }
    atomicAdd(&colsq[b * C_ + tid], a0);
    atomicAdd(&colsq[b * C_ + tid + 256], a1);
}

// ---------------------------------------------------------------------------
// K2b: per-token head-softmax.  16 tokens/block, 128 threads (16 tok x 8 heads)
// ---------------------------------------------------------------------------
#define TOK 16
__global__ __launch_bounds__(128)
void softmax_kernel(const float* __restrict__ W, const float* __restrict__ colsq,
                    const unsigned char* __restrict__ mask, const float* __restrict__ temp,
                    float* __restrict__ Pi, float* __restrict__ sumPi,
                    float* __restrict__ dotAcc)
{
    __shared__ __align__(16) float Wt[TOK][C_];   // 32 KB
    __shared__ __align__(16) float csqInv[C_];
    __shared__ float rowsq[H_][TOK];
    __shared__ float nsum[H_][TOK];
    __shared__ float piL[H_][TOK];
    __shared__ float pdL[H_][TOK];

    const int blk = blockIdx.x;                   // B * N/TOK = 4096 blocks
    const int b   = blk / (N_ / TOK);
    const int n0  = (blk % (N_ / TOK)) * TOK;
    const int tid = threadIdx.x;

    const float* src = W + ((size_t)b * N_ + n0) * C_;
    for (int i = tid; i < TOK * C_ / 4; i += 128)
        ((float4*)&Wt[0][0])[i] = ((const float4*)src)[i];
    for (int i = tid; i < C_; i += 128) {
        const float q = fmaxf(sqrtf(colsq[b * C_ + i]), 1e-12f);
        csqInv[i] = 1.0f / (q * q);
    }
    __syncthreads();

    const int h = tid >> 4;
    const int i = tid & 15;
    float s1 = 0.0f, s2 = 0.0f;
#pragma unroll
    for (int d = 0; d < D_; d++) {
        const float w = Wt[i][h * D_ + d];
        s1 += w * w;
        s2 += w * w * csqInv[h * D_ + d];
    }
    rowsq[h][i] = s1;
    nsum[h][i]  = s2;
    __syncthreads();

    if (tid < TOK) {
        const int n  = n0 + tid;
        const bool m = mask[b * N_ + n] != 0;
        float v[H_], mx = -3.4e38f;
#pragma unroll
        for (int hh = 0; hh < H_; hh++) {
            const float sv = m ? nsum[hh][tid] : -1e9f;
            v[hh] = sv * temp[hh];
            mx = fmaxf(mx, v[hh]);
        }
        float se = 0.0f;
#pragma unroll
        for (int hh = 0; hh < H_; hh++) { v[hh] = __expf(v[hh] - mx); se += v[hh]; }
        const float inv = 1.0f / se;
#pragma unroll
        for (int hh = 0; hh < H_; hh++) {
            const float p = v[hh] * inv;
            Pi[((size_t)(b * H_ + hh)) * N_ + n] = p;
            piL[hh][tid] = p;
            pdL[hh][tid] = p * rowsq[hh][tid];
        }
    }
    __syncthreads();

    if (tid < H_) {
        float sp = 0.0f, sd = 0.0f;
#pragma unroll
        for (int t = 0; t < TOK; t++) { sp += piL[tid][t]; sd += pdL[tid][t]; }
        atomicAdd(&sumPi[b * H_ + tid], sp);
        atomicAdd(&dotAcc[b * H_ + tid], sd);
    }
}

// ---------------------------------------------------------------------------
// K2c: attn[b,h] = 1/(1 + dot/(sumPi+1e-8)); then coef[b,h,n] = -Pi*attn
// ---------------------------------------------------------------------------
__global__ void attn_kernel(const float* __restrict__ sumPi, const float* __restrict__ dotAcc,
                            float* __restrict__ attn)
{
    const int t = threadIdx.x;
    if (t < B_ * H_) {
        const float dots = dotAcc[t] / (sumPi[t] + 1e-8f);
        attn[t] = 1.0f / (1.0f + dots);
    }
}

__global__ __launch_bounds__(256)
void coef_kernel(const float* __restrict__ Pi, const float* __restrict__ attn,
                 float* __restrict__ coef)
{
    const size_t idx = (size_t)blockIdx.x * 256 + threadIdx.x;   // < B*H*N
    const int bh = (int)(idx >> 13);                              // N_ = 8192
    coef[idx] = -Pi[idx] * attn[bh];
}

// ---------------------------------------------------------------------------
// Host-side orchestration
// ---------------------------------------------------------------------------
extern "C" void kernel_launch(void* const* d_in, const int* in_sizes, int n_in,
                              void* d_out, int out_size, void* d_ws, size_t ws_size,
                              hipStream_t stream)
{
    (void)in_sizes; (void)out_size;
    if (n_in < 6) return;

    const float*         x     = (const float*)d_in[0];
    const unsigned char* tmask = (const unsigned char*)d_in[1];
    const float*         Wqkv  = (const float*)d_in[2];
    const float*         temp  = (const float*)d_in[3];
    const float*         Wout  = (const float*)d_in[4];
    const float*         bout  = (const float*)d_in[5];
    float*               out   = (float*)d_out;

    char* ws = (char*)d_ws;
    size_t off = 0;
    auto alloc = [&](size_t bytes) -> void* {
        void* p = ws + off;
        off += (bytes + 255) & ~(size_t)255;
        return p;
    };
    float*          w      = (float*)alloc((size_t)B_ * N_ * C_ * 4);   // 128 MB
    float*          Pi     = (float*)alloc((size_t)B_ * H_ * N_ * 4);   //   2 MB
    float*          coef   = (float*)alloc((size_t)B_ * H_ * N_ * 4);   //   2 MB
    float*          colsq  = (float*)alloc((size_t)B_ * C_ * 4);        //  16 KB
    float*          sums   = (float*)alloc(3 * 64 * 4);                  // sumPi|dot|attn
    unsigned short* WqkvB  = (unsigned short*)alloc((size_t)C_ * C_ * 2);
    unsigned short* WoutB  = (unsigned short*)alloc((size_t)C_ * C_ * 2);
    if (off > ws_size) return;

    float* sumPi  = sums;
    float* dotAcc = sums + 64;
    float* attnv  = sums + 128;

    cvt_bf16_kernel<<<(C_ * C_ + 255) / 256, 256, 0, stream>>>(Wqkv, WqkvB, C_ * C_);
    cvt_bf16_kernel<<<(C_ * C_ + 255) / 256, 256, 0, stream>>>(Wout, WoutB, C_ * C_);
    zero_kernel<<<(B_ * C_ + 192 + 255) / 256, 256, 0, stream>>>(colsq, B_ * C_ + 192);

    dim3 g1(B_ * N_ / 128, C_ / 64);   // (512, 8)
    wmma_gemm_kernel<false><<<g1, 256, 0, stream>>>(x, WqkvB, nullptr, nullptr, w);

    colsq_kernel<<<B_ * N_ / 128, 256, 0, stream>>>(w, colsq);

    softmax_kernel<<<B_ * N_ / TOK, 128, 0, stream>>>(w, colsq, tmask, temp,
                                                      Pi, sumPi, dotAcc);

    attn_kernel<<<1, 64, 0, stream>>>(sumPi, dotAcc, attnv);
    coef_kernel<<<(B_ * H_ * N_) / 256, 256, 0, stream>>>(Pi, attnv, coef);

    wmma_gemm_kernel<true><<<g1, 256, 0, stream>>>(w, WoutB, coef, bout, out);
}
